// MTMADDPGCritic_39831526703501
// MI455X (gfx1250) — compile-verified, matrix operand order checked by reading, not run on hardware
//
#include <hip/hip_runtime.h>
#include <hip/hip_bf16.h>

typedef _Float16 half_t;
typedef _Float16 v8h  __attribute__((ext_vector_type(8)));
typedef _Float16 v16h __attribute__((ext_vector_type(16)));
typedef float    v8f  __attribute__((ext_vector_type(8)));
typedef unsigned int u32x4 __attribute__((ext_vector_type(4)));
typedef int          i32x4 __attribute__((ext_vector_type(4)));
typedef int          i32x8 __attribute__((ext_vector_type(8)));

#define BS 256
#define T_ 100
#define NA 8
#define NE 8
#define N_ACT 14
#define EED 64
#define AED 32
#define HID 256
#define STATE_DIM 224
#define INDIV_DIM 173
#define OBS_DIM 151
#define FEAT_DIM 320
#define NSTEPS (BS * T_)      // 25600 timesteps
#define NROWS (NSTEPS * NA)   // 204800 agent rows
#define RSQRT_AED 0.17677669529663688f

#if __has_builtin(__builtin_amdgcn_tensor_load_to_lds)
#define USE_TDM 1
#endif

// ---------------------------------------------------------------------------
// Kernel P: swizzle W_f1 (320x256 f32) into f16 WMMA B-fragment order.
// Layout: [kt(10)][nsub(16)][lane(32)][e(16)] so a fragment load is one
// contiguous 32B per lane (two global_load_b128, fully coalesced per wave).
// Element e of lane l is K = kt*32 + (e>=8)*16 + (l>=16)*8 + (e&7),
// N = nsub*16 + (l&15)   (ISA 7.12.2 16-bit B layout).
// ---------------------------------------------------------------------------
__global__ void k_swizzle_w(const float* __restrict__ w, half_t* __restrict__ ws) {
    const int i = blockIdx.x * blockDim.x + threadIdx.x;   // 81920 elements
    const int e    = i & 15;
    const int lane = (i >> 4) & 31;
    const int ns   = (i >> 9) & 15;
    const int kt   = i >> 13;
    const int k = kt * 32 + ((e >> 3) << 4) + ((lane >> 4) << 3) + (e & 7);
    const int n = ns * 16 + (lane & 15);
    ws[i] = (half_t)w[k * HID + n];
}

// ---------------------------------------------------------------------------
// Kernel A1: per-timestep entity attention. One wave per (bs,t).
// Writes feat cols [0,64) = ent_emb (broadcast to 8 agents), [64,128) = al_e.
// ---------------------------------------------------------------------------
__global__ __launch_bounds__(128) void k_entity(
    const float* __restrict__ states, const float* __restrict__ ja,
    const float* __restrict__ W_al, const float* __restrict__ b_al,
    const float* __restrict__ W_en, const float* __restrict__ b_en,
    const float* __restrict__ W_sq, const float* __restrict__ b_sq,
    const float* __restrict__ W_sk, const float* __restrict__ b_sk,
    half_t* __restrict__ feat)
{
    __shared__ float ally22[4][8][22];
    __shared__ float ent[4][16][EED];
    __shared__ float qm[4][16][AED];
    __shared__ float km[4][16][AED];
    __shared__ float sc[4][16][16];
    __shared__ float colw[4][16];

    const int w    = threadIdx.x >> 5;
    const int lane = threadIdx.x & 31;
    const int s    = blockIdx.x * 4 + w;
    const float* st = states + (size_t)s * STATE_DIM;
    const float* jp = ja + (size_t)s * (NA * N_ACT);

    // 1) build ally22 = [ally(8), compact(last_act)(7), compact(ja)(7)]
    if (lane < 8) {
        const int a = lane;
        for (int j = 0; j < 8; ++j) ally22[w][a][j] = st[a * 8 + j];
        const float* la = st + 112 + a * 14;
        float sum = 0.f;
        for (int j = 0; j < 6; ++j) ally22[w][a][8 + j] = la[j];
        for (int j = 6; j < 14; ++j) sum += la[j];
        ally22[w][a][14] = sum;
        const float* jaa = jp + a * 14;
        sum = 0.f;
        for (int j = 0; j < 6; ++j) ally22[w][a][15 + j] = jaa[j];
        for (int j = 6; j < 14; ++j) sum += jaa[j];
        ally22[w][a][21] = sum;
    }
    __syncthreads();

    // 2) entity embeddings: rows 0..7 ally, 8..15 enemy. Lane owns 2 columns.
    for (int h = 0; h < 2; ++h) {
        const int c = lane + 32 * h;
        for (int i = 0; i < 8; ++i) {
            float acc = b_al[c];
            for (int j = 0; j < 22; ++j) acc += ally22[w][i][j] * W_al[j * EED + c];
            ent[w][i][c] = acc;
            feat[((size_t)s * 8 + i) * FEAT_DIM + 64 + c] = (half_t)acc;
        }
        for (int e = 0; e < 8; ++e) {
            float acc = b_en[c];
            for (int j = 0; j < 6; ++j) acc += st[64 + e * 6 + j] * W_en[j * EED + c];
            ent[w][8 + e][c] = acc;
        }
    }
    __syncthreads();

    // 3) q,k projections (lane owns one of 32 columns)
    {
        const int c = lane;
        for (int i = 0; i < 16; ++i) {
            float aq = b_sq[c], ak = b_sk[c];
            for (int d = 0; d < EED; ++d) {
                const float e = ent[w][i][d];
                aq += e * W_sq[d * AED + c];
                ak += e * W_sk[d * AED + c];
            }
            qm[w][i][c] = aq;
            km[w][i][c] = ak;
        }
    }
    __syncthreads();

    // 4) scores (256 entries, 8 per lane)
    for (int t = 0; t < 8; ++t) {
        const int idx = lane * 8 + t;
        const int i = idx >> 4, j = idx & 15;
        float acc = 0.f;
        for (int d = 0; d < AED; ++d) acc += qm[w][i][d] * km[w][j][d];
        sc[w][i][j] = acc * RSQRT_AED;
    }
    __syncthreads();

    // 5) row softmax (lanes 0..15)
    if (lane < 16) {
        const int i = lane;
        float mx = sc[w][i][0];
        for (int j = 1; j < 16; ++j) mx = fmaxf(mx, sc[w][i][j]);
        float sum = 0.f;
        for (int j = 0; j < 16; ++j) { float e = __expf(sc[w][i][j] - mx); sc[w][i][j] = e; sum += e; }
        const float inv = 1.f / sum;
        for (int j = 0; j < 16; ++j) sc[w][i][j] *= inv;
    }
    __syncthreads();

    // 6) column weights for mean-pooled attention output
    if (lane < 16) {
        const int j = lane;
        float acc = 0.f;
        for (int i = 0; i < 16; ++i) acc += sc[w][i][j];
        colw[w][j] = acc * (1.0f / 16.0f);
    }
    __syncthreads();

    // 7) ent_emb = sum_j colw[j]*ent[j][:]; broadcast to the 8 agents
    for (int h = 0; h < 2; ++h) {
        const int c = lane + 32 * h;
        float acc = 0.f;
        for (int j = 0; j < 16; ++j) acc += colw[w][j] * ent[w][j][c];
        const half_t hv = (half_t)acc;
        for (int a = 0; a < 8; ++a) feat[((size_t)s * 8 + a) * FEAT_DIM + c] = hv;
    }
}

// ---------------------------------------------------------------------------
// Kernel A2: per-agent-row individual path. One wave per row.
// Writes feat cols [128,192)=own_e, [192,256)=al_emb, [256,320)=en_emb.
// ---------------------------------------------------------------------------
__global__ __launch_bounds__(256) void k_indiv(
    const float* __restrict__ x_all,
    const float* __restrict__ W_oq, const float* __restrict__ b_oq,
    const float* __restrict__ W_ak, const float* __restrict__ b_ak,
    const float* __restrict__ W_av, const float* __restrict__ b_av,
    const float* __restrict__ W_ek, const float* __restrict__ b_ek,
    const float* __restrict__ W_ev, const float* __restrict__ b_ev,
    const float* __restrict__ W_ov, const float* __restrict__ b_ov,
    half_t* __restrict__ feat)
{
    __shared__ float own[8][32];
    __shared__ float oq[8][32];
    __shared__ float kk[8][8][32];
    __shared__ float ww[8][8];

    const int w    = threadIdx.x >> 5;
    const int lane = threadIdx.x & 31;
    const size_t r = (size_t)blockIdx.x * 8 + w;
    const int a = (int)(r & 7);
    const float* x  = x_all + r * INDIV_DIM;
    const float* la = x + OBS_DIM;

    // own31 = [own(16), agent_id(8), compact(last_act)(7)]
    {
        float v = 0.f;
        if (lane < 16)      v = x[lane];
        else if (lane < 24) v = (float)(((a + 1) >> (7 - (lane - 16))) & 1);
        else if (lane < 30) v = la[lane - 24];
        else if (lane == 30) { for (int j = 6; j < 14; ++j) v += la[j]; }
        own[w][lane] = (lane < 31) ? v : 0.f;
    }
    __syncthreads();

    // own_e (2 cols/lane) and oq (1 col/lane)
    {
        float aq = b_oq[lane];
        float a0 = b_ov[lane], a1 = b_ov[lane + 32];
        for (int j = 0; j < 31; ++j) {
            const float ov = own[w][j];
            aq += ov * W_oq[j * AED + lane];
            a0 += ov * W_ov[j * EED + lane];
            a1 += ov * W_ov[j * EED + lane + 32];
        }
        oq[w][lane] = aq;
        feat[r * FEAT_DIM + 128 + lane]      = (half_t)a0;
        feat[r * FEAT_DIM + 128 + lane + 32] = (half_t)a1;
    }
    __syncthreads();

    // ---- ally attention: 7 keys, 9 input feats ----
    const float* alf = x + 16 + NE * 9;   // offset 88
    for (int j = 0; j < 7; ++j) {
        float acc = b_ak[lane];
        const float* f = alf + j * 9;
        for (int t = 0; t < 9; ++t) acc += f[t] * W_ak[t * AED + lane];
        kk[w][j][lane] = acc;
    }
    __syncthreads();
    if (lane == 0) {
        float s[7];
        float mx = -1e30f;
        for (int j = 0; j < 7; ++j) {
            float acc = 0.f;
            for (int c = 0; c < 32; ++c) acc += oq[w][c] * kk[w][j][c];
            s[j] = acc * RSQRT_AED;
            mx = fmaxf(mx, s[j]);
        }
        float sum = 0.f;
        for (int j = 0; j < 7; ++j) { s[j] = __expf(s[j] - mx); sum += s[j]; }
        const float inv = 1.f / sum;
        for (int j = 0; j < 7; ++j) ww[w][j] = s[j] * inv;
    }
    __syncthreads();
    {
        float a0 = 0.f, a1 = 0.f;
        for (int j = 0; j < 7; ++j) {
            const float* f = alf + j * 9;
            float v0 = b_av[lane], v1 = b_av[lane + 32];
            for (int t = 0; t < 9; ++t) { const float fv = f[t]; v0 += fv * W_av[t * EED + lane]; v1 += fv * W_av[t * EED + lane + 32]; }
            const float wj = ww[w][j];
            a0 += wj * v0; a1 += wj * v1;
        }
        feat[r * FEAT_DIM + 192 + lane]      = (half_t)a0;
        feat[r * FEAT_DIM + 192 + lane + 32] = (half_t)a1;
    }
    __syncthreads();

    // ---- enemy attention: 8 keys, 10 input feats (9 obs + attack) ----
    const float* enf = x + 16;
    for (int j = 0; j < 8; ++j) {
        float acc = b_ek[lane];
        const float* f = enf + j * 9;
        for (int t = 0; t < 9; ++t) acc += f[t] * W_ek[t * AED + lane];
        acc += la[6 + j] * W_ek[9 * AED + lane];
        kk[w][j][lane] = acc;
    }
    __syncthreads();
    if (lane == 0) {
        float s[8];
        float mx = -1e30f;
        for (int j = 0; j < 8; ++j) {
            float acc = 0.f;
            for (int c = 0; c < 32; ++c) acc += oq[w][c] * kk[w][j][c];
            s[j] = acc * RSQRT_AED;
            mx = fmaxf(mx, s[j]);
        }
        float sum = 0.f;
        for (int j = 0; j < 8; ++j) { s[j] = __expf(s[j] - mx); sum += s[j]; }
        const float inv = 1.f / sum;
        for (int j = 0; j < 8; ++j) ww[w][j] = s[j] * inv;
    }
    __syncthreads();
    {
        float a0 = 0.f, a1 = 0.f;
        for (int j = 0; j < 8; ++j) {
            const float* f = enf + j * 9;
            float v0 = b_ev[lane], v1 = b_ev[lane + 32];
            for (int t = 0; t < 9; ++t) { const float fv = f[t]; v0 += fv * W_ev[t * EED + lane]; v1 += fv * W_ev[t * EED + lane + 32]; }
            const float atk = la[6 + j];
            v0 += atk * W_ev[9 * EED + lane];
            v1 += atk * W_ev[9 * EED + lane + 32];
            const float wj = ww[w][j];
            a0 += wj * v0; a1 += wj * v1;
        }
        feat[r * FEAT_DIM + 256 + lane]      = (half_t)a0;
        feat[r * FEAT_DIM + 256 + lane + 32] = (half_t)a1;
    }
}

// ---------------------------------------------------------------------------
// Kernel B: WMMA GEMM  out = relu(feat @ W_f1 + b_f1) @ W_f2 + b_f2
// Block = 256 threads (8 waves), M-tile 64 x N 256, K = 320 in 10 steps.
// A slab staged to LDS by the Tensor Data Mover (double-buffered); B frags
// loaded directly from the pre-swizzled global tensor (L2-resident).
// ---------------------------------------------------------------------------
#define APAD 40   // LDS row stride in halfs (80B); TDM pad: 16B after every 64B

__global__ __launch_bounds__(256) void k_gemm(
    const half_t* __restrict__ feat, const half_t* __restrict__ w1s,
    const float* __restrict__ b1, const float* __restrict__ w2,
    const float* __restrict__ b2, float* __restrict__ out)
{
    __shared__ half_t As[2][64 * APAD];
    __shared__ float rowsum[64];

    const int tid  = threadIdx.x;
    const int w    = tid >> 5;
    const int lane = tid & 31;
    const int wave_m = w >> 2;      // 0..1
    const int wave_n = w & 3;       // 0..3
    const size_t m0 = (size_t)blockIdx.x * 64;

    if (tid < 64) rowsum[tid] = b2[0];

    v8f acc[2][4];
    for (int i = 0; i < 2; ++i)
        for (int j = 0; j < 4; ++j)
            for (int q = 0; q < 8; ++q) acc[i][j][q] = 0.0f;

    const int b0   = (lane >= 16) ? 8 : 0;   // K-group offset per ISA A/B layout
    const int nrow = lane & 15;

#ifdef USE_TDM
    // Tensor DMA descriptor group 1 (constant): data_size=2B, pad_enable,
    // pad_interval=3 (pad after 16 DWORDs = 64B), pad_amount=3 (4 DWORDs =
    // 16B) -> LDS row stride 80B = APAD halfs. Tile 32x64, row stride 320.
    i32x8 g1;
    g1[0] = (1 << 16) | (1 << 20) | (3 << 22) | (3 << 25);
    g1[1] = (int)0xFFFF0000u;            // tensor_dim0[15:0]=0xFFFF (huge)
    g1[2] = (int)0xFFFF7FFFu;            // tensor_dim0[31:16], tensor_dim1[15:0]
    g1[3] = (int)((32u << 16) | 0x7FFFu);// tensor_dim1[31:16], tile_dim0=32
    g1[4] = 64;                          // tile_dim1=64, tile_dim2=0
    g1[5] = FEAT_DIM;                    // tensor_dim0_stride = 320 elements
    g1[6] = 0;
    g1[7] = 0;
    i32x4 gz4;
    gz4[0] = 0; gz4[1] = 0; gz4[2] = 0; gz4[3] = 0;
    i32x8 gz8;
    for (int q = 0; q < 8; ++q) gz8[q] = 0;

    const unsigned lds0 = (unsigned)(size_t)(&As[0][0]);
    const unsigned lds1 = (unsigned)(size_t)(&As[1][0]);

#define ISSUE_TDM(KT, BUF)                                                    \
    do {                                                                      \
        unsigned long long ga =                                               \
            (unsigned long long)(size_t)(const void*)(feat + m0 * FEAT_DIM + (KT) * 32); \
        u32x4 g0;                                                             \
        g0[0] = 1u;                      /* count=1 valid descriptor */       \
        g0[1] = (BUF) ? lds1 : lds0;     /* lds_addr */                       \
        g0[2] = (unsigned)(ga & 0xFFFFFFFFu);                                 \
        g0[3] = (unsigned)((ga >> 32) & 0x1FFFFFFu) | (2u << 30); /* type=2 */\
        __builtin_amdgcn_tensor_load_to_lds(g0, g1, gz4, gz4, gz8, 0);        \
    } while (0)

    if (w == 0) ISSUE_TDM(0, 0);
#endif

    for (int kt = 0; kt < 10; ++kt) {
#ifdef USE_TDM
        if (w == 0) {
            if (kt < 9) {
                ISSUE_TDM(kt + 1, (kt + 1) & 1);
                __builtin_amdgcn_s_wait_tensorcnt(1);   // tile kt complete
            } else {
                __builtin_amdgcn_s_wait_tensorcnt(0);
            }
        }
#else
        {   // manual fallback staging: 64 rows x 32 halfs, one b128/thread
            const int row = tid >> 2, chunk = tid & 3;
            const v8h* src = (const v8h*)(feat + (m0 + row) * FEAT_DIM + kt * 32 + chunk * 8);
            *(v8h*)(&As[kt & 1][row * APAD + chunk * 8]) = *src;
        }
#endif
        __syncthreads();

        // A fragments (ISA 16-bit A layout: halfs K=[b0,b0+8) and [b0+16,b0+24))
        const half_t* abuf = &As[kt & 1][0];
        v16h afrag[2];
        for (int ms = 0; ms < 2; ++ms) {
            const int rrow = (wave_m * 2 + ms) * 16 + nrow;
            const v8h lo = *(const v8h*)(abuf + rrow * APAD + b0);
            const v8h hi = *(const v8h*)(abuf + rrow * APAD + b0 + 16);
            v16h av;
            for (int q = 0; q < 8; ++q) { av[q] = lo[q]; av[q + 8] = hi[q]; }
            afrag[ms] = av;
        }
        // B fragments straight from pre-swizzled global (32B/lane, coalesced)
        for (int ns = 0; ns < 4; ++ns) {
            const half_t* bb = w1s + ((size_t)(kt * 16 + wave_n * 4 + ns) * 32 + lane) * 16;
            const v16h bv = *(const v16h*)bb;
            for (int ms = 0; ms < 2; ++ms) {
                acc[ms][ns] = __builtin_amdgcn_wmma_f32_16x16x32_f16(
                    false, afrag[ms], false, bv, (short)0, acc[ms][ns], false, false);
            }
        }
        __syncthreads();
    }

    // fused epilogue: relu(h + b1) dot W_f2, reduced per output row
    float bias[4], wv[4];
    for (int ns = 0; ns < 4; ++ns) {
        const int n = (wave_n * 4 + ns) * 16 + nrow;
        bias[ns] = b1[n];
        wv[ns]   = w2[n];
    }
    const int rhalf = (lane >= 16) ? 8 : 0;
    for (int ms = 0; ms < 2; ++ms) {
#pragma unroll
        for (int rr = 0; rr < 8; ++rr) {
            float p = 0.f;
#pragma unroll
            for (int ns = 0; ns < 4; ++ns) {
                float h = acc[ms][ns][rr] + bias[ns];
                h = fmaxf(h, 0.f);
                p += h * wv[ns];
            }
            // reduce across the 16 lanes of each half-wave (same row, diff cols)
            p += __shfl_xor(p, 1, 16);
            p += __shfl_xor(p, 2, 16);
            p += __shfl_xor(p, 4, 16);
            p += __shfl_xor(p, 8, 16);
            if ((lane & 15) == 0) {
                const int rl = (wave_m * 2 + ms) * 16 + rr + rhalf;
                atomicAdd(&rowsum[rl], p);
            }
        }
    }
    __syncthreads();
    if (tid < 64) out[m0 + tid] = rowsum[tid];
}

// ---------------------------------------------------------------------------
// Launch
// ---------------------------------------------------------------------------
extern "C" void kernel_launch(void* const* d_in, const int* in_sizes, int n_in,
                              void* d_out, int out_size, void* d_ws, size_t ws_size,
                              hipStream_t stream) {
    const float* states = (const float*)d_in[0];
    const float* ja     = (const float*)d_in[1];
    const float* indiv  = (const float*)d_in[2];
    const float* W_al = (const float*)d_in[3];  const float* b_al = (const float*)d_in[4];
    const float* W_en = (const float*)d_in[5];  const float* b_en = (const float*)d_in[6];
    const float* W_sq = (const float*)d_in[7];  const float* b_sq = (const float*)d_in[8];
    const float* W_sk = (const float*)d_in[9];  const float* b_sk = (const float*)d_in[10];
    const float* W_oq = (const float*)d_in[11]; const float* b_oq = (const float*)d_in[12];
    const float* W_ak = (const float*)d_in[13]; const float* b_ak = (const float*)d_in[14];
    const float* W_av = (const float*)d_in[15]; const float* b_av = (const float*)d_in[16];
    const float* W_ek = (const float*)d_in[17]; const float* b_ek = (const float*)d_in[18];
    const float* W_ev = (const float*)d_in[19]; const float* b_ev = (const float*)d_in[20];
    const float* W_ov = (const float*)d_in[21]; const float* b_ov = (const float*)d_in[22];
    const float* W_f1 = (const float*)d_in[23]; const float* b_f1 = (const float*)d_in[24];
    const float* W_f2 = (const float*)d_in[25]; const float* b_f2 = (const float*)d_in[26];

    // workspace: feat f16 (204800 x 320) then swizzled W_f1 f16 (320 x 256)
    half_t* feat = (half_t*)d_ws;
    half_t* w1s  = (half_t*)((char*)d_ws + (size_t)NROWS * FEAT_DIM * sizeof(half_t));

    k_swizzle_w<<<(FEAT_DIM * HID + 255) / 256, 256, 0, stream>>>(W_f1, w1s);

    k_entity<<<NSTEPS / 4, 128, 0, stream>>>(states, ja,
        W_al, b_al, W_en, b_en, W_sq, b_sq, W_sk, b_sk, feat);

    k_indiv<<<NROWS / 8, 256, 0, stream>>>(indiv,
        W_oq, b_oq, W_ak, b_ak, W_av, b_av, W_ek, b_ek, W_ev, b_ev, W_ov, b_ov, feat);

    k_gemm<<<NROWS / 64, 256, 0, stream>>>(feat, w1s, b_f1, W_f2, b_f2, (float*)d_out);
}
